// ArcFace_83700322664537
// MI455X (gfx1250) — matrix operand order repaired and checked by reading, not compiled, not run on hardware
//
#include <hip/hip_runtime.h>

// Problem constants (reference: N=512, D=512, C=100000, S=1)
#define NB   512
#define DD   512
#define CCLS 100000
#define CT   32        // classes per block
#define KC   32        // K (embedding dim) chunk per WMMA step
#define LGS  513       // padded LDS row stride for logits (513 % 64 == 1 -> conflict free)

typedef __attribute__((ext_vector_type(16))) __bf16 v16bf;
typedef __attribute__((ext_vector_type(8)))  float  v8f;

__device__ __forceinline__ v8f wmma_bf16(v16bf a, v16bf b, v8f c) {
  // D = A(16x32 bf16) * B(32x16 bf16) + C(16x16 f32)
  return __builtin_amdgcn_wmma_f32_16x16x32_bf16(
      /*neg_a=*/false, a, /*neg_b=*/false, b,
      /*c_mod=*/(short)0, c, /*reuse_a=*/false, /*reuse_b=*/false);
}

// ---------------------------------------------------------------------------
// Prep: convert embeddings f32 -> bf16 into WMMA A-fragment swizzled layout.
// Flat index i = ((mt*16 + kt)*32 + lane)*16 + e  maps to
//   row = mt*16 + (lane%16)
//   K   = kt*32 + (e/8)*16 + (lane/16)*8 + (e%8)     (ISA 16-bit A 16x32 layout)
// so in the GEMM each lane loads 32 contiguous bytes (two global_load_b128).
// ---------------------------------------------------------------------------
__global__ void arcface_prep_emb(const float* __restrict__ emb,
                                 __bf16* __restrict__ eswz) {
  int i    = blockIdx.x * 256 + threadIdx.x;   // 0 .. 512*512-1
  int e    =  i        & 15;
  int lane = (i >> 4)  & 31;
  int kt   = (i >> 9)  & 15;
  int mtg  =  i >> 13;                         // 0..31
  int row  = mtg * 16 + (lane & 15);
  int k    = kt * KC + ((e >> 3) << 4) + ((lane >> 4) << 3) + (e & 7);
  eswz[i] = (__bf16)emb[row * DD + k];
}

// ---------------------------------------------------------------------------
// Fused: normalized-GEMM (bf16 WMMA, f32 accum) + ArcFace margin + softmax
// over the batch axis. One block = 32 classes x all 512 batch rows.
// ---------------------------------------------------------------------------
__global__ __launch_bounds__(256) void arcface_main(
    const float* __restrict__ w,        // [D, C] row-major
    const float* __restrict__ target,   // [N, C] one-hot
    const v16bf* __restrict__ eswz,     // swizzled bf16 embeddings
    float* __restrict__ out) {          // [N, C]
  __shared__ v16bf  Bsh[2][32];         // two 32x16 bf16 B fragments (2 KB)
  __shared__ float  Lg[CT * LGS];       // logits, padded (64.1 KB)
  __shared__ float4 red4[256];          // reduction scratch (4 KB)
  __shared__ float  rinv[CT];
  __shared__ float  maxv[CT];
  __shared__ float  rsum[CT];

  const float COSM = 0.8775825618903728f;  // cos(0.5)
  const float SINM = 0.479425538604203f;   // sin(0.5)

  const int tid  = threadIdx.x;
  const int wid  = tid >> 5;     // wave 0..7 -> rows [wid*64, wid*64+64)
  const int lane = tid & 31;
  const int l16  = lane & 15;
  const int hi8  = (lane >> 4) << 3;       // C/D layout: M offset 0 or 8
  const int c0   = blockIdx.x * CT;

  // w-tile loader mapping: thread -> (k row within chunk, 4 classes)
  const int kk  = tid >> 3;      // 0..31
  const int cc4 = tid & 7;       // 0..7  -> classes cc4*4 .. cc4*4+3
  const float* wp = w + (size_t)kk * CCLS + c0 + cc4 * 4;

  v8f acc[4][2];
#pragma unroll
  for (int mt = 0; mt < 4; ++mt)
#pragma unroll
    for (int nt = 0; nt < 2; ++nt)
      acc[mt][nt] = (v8f){0.f, 0.f, 0.f, 0.f, 0.f, 0.f, 0.f, 0.f};

  float sq0 = 0.f, sq1 = 0.f, sq2 = 0.f, sq3 = 0.f;

  // ---------------- K loop: D=512 in 16 chunks of 32 ----------------
  for (int kt = 0; kt < 16; ++kt) {
    const float4 wv = *(const float4*)(wp + (size_t)kt * KC * CCLS);
    if (kt < 15)
      __builtin_prefetch(wp + (size_t)(kt + 1) * KC * CCLS, 0, 0);
    sq0 += wv.x * wv.x; sq1 += wv.y * wv.y;
    sq2 += wv.z * wv.z; sq3 += wv.w * wv.w;

    __syncthreads();   // previous WMMA B reads complete
    {
      __bf16* bs = (__bf16*)Bsh;
      const float wa[4] = {wv.x, wv.y, wv.z, wv.w};
#pragma unroll
      for (int q = 0; q < 4; ++q) {
        const int cl = cc4 * 4 + q;            // class 0..31 in tile
        const int nt = cl >> 4;
        const int bl = (cl & 15) + ((kk & 8) << 1);  // B lane: col + 16*((K/8)&1)
        bs[nt * 512 + bl * 16 + (kk & 7) + ((kk & 16) >> 1)] = (__bf16)wa[q];
      }
    }
    __syncthreads();   // B fragment visible

#pragma unroll
    for (int mt = 0; mt < 4; ++mt) {
      const v16bf a = eswz[((wid * 4 + mt) * 16 + kt) * 32 + lane];
      acc[mt][0] = wmma_bf16(a, Bsh[0][lane], acc[mt][0]);
      acc[mt][1] = wmma_bf16(a, Bsh[1][lane], acc[mt][1]);
    }
  }

  // ---------------- per-class 1/||w_c|| from streamed tile ----------------
  red4[tid] = make_float4(sq0, sq1, sq2, sq3);
  __syncthreads();
  if (tid < 8) {
    float s0 = 0.f, s1 = 0.f, s2 = 0.f, s3 = 0.f;
    for (int j = 0; j < 32; ++j) {
      const float4 v = red4[j * 8 + tid];
      s0 += v.x; s1 += v.y; s2 += v.z; s3 += v.w;
    }
    rinv[tid * 4 + 0] = rsqrtf(s0);
    rinv[tid * 4 + 1] = rsqrtf(s1);
    rinv[tid * 4 + 2] = rsqrtf(s2);
    rinv[tid * 4 + 3] = rsqrtf(s3);
  }
  __syncthreads();

  // ---------------- epilogue: clamp, margin, scale -> logits in LDS -------
#pragma unroll
  for (int mt = 0; mt < 4; ++mt) {
#pragma unroll
    for (int nt = 0; nt < 2; ++nt) {
      const int   c  = c0 + nt * 16 + l16;
      const float ri = rinv[nt * 16 + l16];
#pragma unroll
      for (int r = 0; r < 8; ++r) {
        const int row = wid * 64 + mt * 16 + r + hi8;
        float x = acc[mt][nt][r] * ri;
        x = fminf(1.f, fmaxf(-1.f, x));
        const float t  = target[(size_t)row * CCLS + c];  // 0 or 1
        const float cm = 1.f + t * (COSM - 1.f);
        const float sm = t * SINM;
        const float s  = sqrtf(fmaxf(0.f, 1.f - x * x));  // sin(acos(x))
        Lg[(nt * 16 + l16) * LGS + row] = 64.f * (x * cm - s * sm);
      }
    }
  }
  __syncthreads();

  // ---------------- softmax over batch axis (512 rows per class) ----------
  float* redf = (float*)red4;
  {
    const int cl = tid >> 3, seg = tid & 7;
    const float* col = &Lg[cl * LGS + seg * 64];
    float m = -3.0e38f;
    for (int i = 0; i < 64; ++i) m = fmaxf(m, col[i]);
    redf[tid] = m;
  }
  __syncthreads();
  if (tid < CT) {
    float m = -3.0e38f;
    for (int j = 0; j < 8; ++j) m = fmaxf(m, redf[tid * 8 + j]);
    maxv[tid] = m;
  }
  __syncthreads();
  {
    const int cl = tid >> 3, seg = tid & 7;
    const float mv = maxv[cl];
    const float* col = &Lg[cl * LGS + seg * 64];
    float s = 0.f;
    for (int i = 0; i < 64; ++i) s += __expf(col[i] - mv);
    redf[tid] = s;
  }
  __syncthreads();
  if (tid < CT) {
    float s = 0.f;
    for (int j = 0; j < 8; ++j) s += redf[tid * 8 + j];
    rsum[tid] = 1.f / s;
  }
  __syncthreads();

  // coalesced writes: 32 consecutive classes per wave row
  {
    const int cl = tid & 31;
    const int rb = tid >> 5;
    const float mv = maxv[cl];
    const float rs = rsum[cl];
    for (int it = 0; it < 64; ++it) {
      const int row = it * 8 + rb;
      out[(size_t)row * CCLS + c0 + cl] =
          __expf(Lg[cl * LGS + row] - mv) * rs;
    }
  }
}

extern "C" void kernel_launch(void* const* d_in, const int* in_sizes, int n_in,
                              void* d_out, int out_size, void* d_ws, size_t ws_size,
                              hipStream_t stream) {
  const float* emb    = (const float*)d_in[0];   // [512, 512]
  const float* w      = (const float*)d_in[1];   // [512, 100000, 1]
  const float* target = (const float*)d_in[2];   // [512, 100000]
  float*       out    = (float*)d_out;           // [512, 100000]

  __bf16* eswz = (__bf16*)d_ws;                  // 512 KB swizzled bf16 emb

  arcface_prep_emb<<<(NB * DD) / 256, 256, 0, stream>>>(emb, eswz);
  arcface_main<<<CCLS / CT, 256, 0, stream>>>(w, target, (const v16bf*)eswz, out);
}
